// EdgeGroupAnalyzer_34256659153226
// MI455X (gfx1250) — compile-verified, matrix operand order checked by reading, not run on hardware
//
#include <hip/hip_runtime.h>
#include <hip/hip_bf16.h>

typedef __attribute__((ext_vector_type(16))) __bf16 v16bf;
typedef __attribute__((ext_vector_type(8)))  float  v8f;
typedef __attribute__((ext_vector_type(4)))  __bf16 bf16x4;

#define N_NODES 50000
#define E_EDGES 800000
#define DIM     256
#define K_TOT   512      // 2*DIM
#define TOPK    50
#define M_TILE  32       // edges per workgroup tile

// LDS layout (all fragment-ordered: one WMMA operand = 1024 B, lane l owns [32l,32l+32))
#define W1T_BYTES   (256u * K_TOT * 2u)              // 262144 B  (16 N-tiles x 16 K-steps)
#define A_BUF_BYTES (32u * K_TOT * 2u)               // 32768 B   (2 M-tiles x 16 K-steps)
#define SMEM_BYTES  (W1T_BYTES + 2u * A_BUF_BYTES)   // 327680 B == 320 KB exactly

// workspace layout
#define WS_EMB16_BYTES ((size_t)N_NODES * DIM * 2)   // 25,600,000
#define WS_W1T_OFF     WS_EMB16_BYTES
#define WS_SCORES_OFF  (WS_EMB16_BYTES + W1T_BYTES)

// ---------------------------------------------------------------------------
// Kernel 0a: W1 (512x256 f32 row-major) -> bf16 W1^T in WMMA fragment order.
// dest elem index for (k,n):  frag = (n/16)*16 + k/32 ; lane = (n%16) + 16*((k%32)/16)
//                             dst  = frag*512 + lane*16 + (k%16)
// ---------------------------------------------------------------------------
__global__ void k_convert_w1(const float* __restrict__ W1, __bf16* __restrict__ w1t) {
    int idx = blockIdx.x * 256 + threadIdx.x;
    if (idx < K_TOT * DIM) {
        int k = idx >> 8;    // 0..511
        int n = idx & 255;   // 0..255
        int frag = ((n >> 4) << 4) + (k >> 5);
        int l    = (n & 15) | (((k >> 4) & 1) << 4);
        int dst  = frag * 512 + l * 16 + (k & 15);
        w1t[dst] = (__bf16)W1[idx];
    }
}

// ---------------------------------------------------------------------------
// Kernel 0b: embeddings f32 -> bf16 table (enables async LDS gather, halves
// L2 gather traffic)
// ---------------------------------------------------------------------------
__global__ void k_convert_emb(const float* __restrict__ emb, __bf16* __restrict__ emb16) {
    int i = blockIdx.x * 256 + threadIdx.x;          // one float4 per thread
    if (i < (N_NODES * DIM) / 4) {
        float4 v = ((const float4*)emb)[i];
        bf16x4 b;
        b[0] = (__bf16)v.x; b[1] = (__bf16)v.y;
        b[2] = (__bf16)v.z; b[3] = (__bf16)v.w;
        ((bf16x4*)emb16)[i] = b;
    }
}

// ---------------------------------------------------------------------------
// Kernel 1: persistent fused gather(async)+GEMM(bf16 WMMA)+bias/ReLU/W2/sigmoid
// 256 threads = 8 wave32s, compute layout 2(M) x 4(N), double-buffered A tile,
// software-pipelined K loop.
// ---------------------------------------------------------------------------
__global__ void __launch_bounds__(256)
k_edge_score(const __bf16*    __restrict__ emb16,
             const __bf16*    __restrict__ w1t_g,
             const float*     __restrict__ b1,
             const float*     __restrict__ w2,
             const float*     __restrict__ b2,
             const long long* __restrict__ eidx,
             float*           __restrict__ scores,
             int n_tiles) {
    extern __shared__ char smem[];
    char* w1t   = smem;                               // 256 KB, fragment order
    char* abuf0 = smem + W1T_BYTES;                   // 2 x 32 KB ping-pong

    const int tid   = threadIdx.x;
    const int lane  = tid & 31;
    const int wave  = tid >> 5;
    const int waveM = wave & 1;   // M rows [waveM*16, +16)
    const int waveN = wave >> 1;  // N cols [waveN*64, +64)

    // ---- async-gather issue for one 32-edge tile into `buf` --------------
    // wave w, lane l owns edge-row r=(w>>2)*16+(l&15), k-quadrant q=w&3.
    auto prefetch_tile = [&](int tile, char* buf) {
        const int mt = wave >> 2;                 // 0..1  (M sub-tile)
        const int q  = wave & 3;                  // 0..3  (128 k-cols each)
        const int r  = mt * 16 + (lane & 15);
        const int k0 = q * 128 + ((lane >> 4) << 4);
        long long e    = (long long)tile * M_TILE + r;
        long long node = eidx[(long long)(k0 >= DIM ? E_EDGES : 0) + e];
        unsigned long long ga =
            (unsigned long long)(uintptr_t)(emb16 + node * DIM + (k0 & (DIM - 1)));
        unsigned int lds0 =
            (unsigned int)(uintptr_t)buf + (unsigned)((mt * 16 + q * 4) << 10) + lane * 32;
        #pragma unroll
        for (int j = 0; j < 4; ++j) {             // 4 K-step fragments
            unsigned int la = lds0 + (unsigned)(j << 10);
            unsigned long long gj = ga + (unsigned long long)(j * 64);
            // same 16B offset applies to both LDS and global sides (ISA 10.7)
            asm volatile("global_load_async_to_lds_b128 %0, %1, off"
                         :: "v"(la), "v"(gj) : "memory");
            asm volatile("global_load_async_to_lds_b128 %0, %1, off offset:16"
                         :: "v"(la), "v"(gj) : "memory");
        }
    };

    int tile = blockIdx.x;
    if (tile < n_tiles) prefetch_tile(tile, abuf0);   // prologue fill of buf0

    // ---- stage full W1^T into LDS (linear copy, already fragment order) --
    for (int i = tid; i < (int)(W1T_BYTES / 16); i += 256)
        ((float4*)w1t)[i] = ((const float4*)w1t_g)[i];

    unsigned cur_off = 0;                             // XOR ping-pong offset
    for (; tile < n_tiles; tile += gridDim.x, cur_off ^= A_BUF_BYTES) {
        char* cur = abuf0 + cur_off;
        char* nxt = abuf0 + (cur_off ^ A_BUF_BYTES);

        asm volatile("s_wait_asynccnt 0x0" ::: "memory");  // cur buffer filled
        __syncthreads();                                   // visible to all waves

        int tnext = tile + gridDim.x;
        if (tnext < n_tiles) prefetch_tile(tnext, nxt);    // overlap with K loop

        // ---- software-pipelined K loop: 16 steps x 4 N-tiles of
        //      wmma_f32_16x16x32_bf16; stage ks+1 loads before ks WMMAs -----
        const char* Abase = cur + ((waveM * 16) << 10) + lane * 32;
        const char* Bbase = w1t + (((waveN * 4) * 16) << 10) + lane * 32;

        v8f acc[4] = {};
        v16bf a_cur = *(const v16bf*)(Abase);
        v16bf b_cur[4];
        #pragma unroll
        for (int nt = 0; nt < 4; ++nt)
            b_cur[nt] = *(const v16bf*)(Bbase + ((nt * 16) << 10));

        #pragma unroll
        for (int ks = 0; ks < 16; ++ks) {
            v16bf a_nxt = a_cur;
            v16bf b_nxt[4];
            if (ks < 15) {
                a_nxt = *(const v16bf*)(Abase + ((ks + 1) << 10));
                #pragma unroll
                for (int nt = 0; nt < 4; ++nt)
                    b_nxt[nt] = *(const v16bf*)(Bbase + ((nt * 16 + ks + 1) << 10));
            }
            #pragma unroll
            for (int nt = 0; nt < 4; ++nt)
                acc[nt] = __builtin_amdgcn_wmma_f32_16x16x32_bf16(
                    false, a_cur, false, b_cur[nt], (short)0, acc[nt], false, false);
            a_cur = a_nxt;
            #pragma unroll
            for (int nt = 0; nt < 4; ++nt) b_cur[nt] = (ks < 15) ? b_nxt[nt] : b_cur[nt];
        }
        __syncthreads();                 // all reads of cur done -> cur is dead

        float* ssc = (float*)cur;        // reuse dead A buffer as reduce scratch
        if (tid < M_TILE) ssc[tid] = 0.0f;
        __syncthreads();

        // ---- epilogue: bias + ReLU + dot with W2 -------------------------
        float p[8];
        #pragma unroll
        for (int v = 0; v < 8; ++v) p[v] = 0.0f;
        #pragma unroll
        for (int nt = 0; nt < 4; ++nt) {
            int n = waveN * 64 + nt * 16 + (lane & 15);
            float bb = b1[n];
            float ww = w2[n];
            #pragma unroll
            for (int v = 0; v < 8; ++v) {
                float h = acc[nt][v] + bb;
                h = fmaxf(h, 0.0f);
                p[v] += h * ww;
            }
        }
        #pragma unroll
        for (int off = 8; off >= 1; off >>= 1) {
            #pragma unroll
            for (int v = 0; v < 8; ++v) p[v] += __shfl_xor(p[v], off, 16);
        }
        if ((lane & 15) == 0) {
            int mbase = waveM * 16 + ((lane >> 4) ? 8 : 0);
            #pragma unroll
            for (int v = 0; v < 8; ++v) atomicAdd(&ssc[mbase + v], p[v]);
        }
        __syncthreads();

        if (tid < M_TILE) {
            long long e = (long long)tile * M_TILE + tid;
            long long s = eidx[e];
            long long d = eidx[E_EDGES + e];
            float x  = ssc[tid] + b2[0];
            float sc = 1.0f / (1.0f + __expf(-x));
            scores[e] = (s < d) ? sc : -1.0f;
        }
        // next-iteration head (s_wait_asynccnt + barrier) orders the refill of
        // `cur` after this finalize.
    }
}

// ---------------------------------------------------------------------------
// Kernel 2: top-50 via 50 rounds of partitioned argmax (single block).
// Element e is owned by thread e%256, so the exclusion write scores[sel]=-4
// is always read back by its own writer (no cross-thread cache hazard).
// ---------------------------------------------------------------------------
__global__ void k_topk(float* __restrict__ scores,
                       const long long* __restrict__ eidx,
                       float* __restrict__ out) {
    __shared__ float bval[256];
    __shared__ int   bidx[256];
    __shared__ int   sel_s;
    const int tid = threadIdx.x;

    for (int r = 0; r < TOPK; ++r) {
        float best = -1e30f;
        int   bi   = 0;
        for (int e = tid; e < E_EDGES; e += 256) {
            float s = scores[e];
            if (s > best) { best = s; bi = e; }
        }
        bval[tid] = best;
        bidx[tid] = bi;
        __syncthreads();
        if (tid == 0) {
            float bb = bval[0]; int bbi = bidx[0];
            for (int t = 1; t < 256; ++t)
                if (bval[t] > bb) { bb = bval[t]; bbi = bidx[t]; }
            sel_s = bbi;
            out[r]            = (float)eidx[bbi];             // top_pairs row 0
            out[TOPK + r]     = (float)eidx[E_EDGES + bbi];   // top_pairs row 1
            out[2 * TOPK + r] = bb;                           // top_scores
        }
        __syncthreads();
        int sel = sel_s;
        if ((sel & 255) == tid) scores[sel] = -4.0f;
        __syncthreads();
    }
}

// ---------------------------------------------------------------------------
extern "C" void kernel_launch(void* const* d_in, const int* in_sizes, int n_in,
                              void* d_out, int out_size, void* d_ws, size_t ws_size,
                              hipStream_t stream) {
    const float*     emb  = (const float*)d_in[0];      // (50000,256) f32
    const float*     W1   = (const float*)d_in[1];      // (512,256)   f32
    const float*     b1   = (const float*)d_in[2];      // (256,)      f32
    const float*     W2   = (const float*)d_in[3];      // (256,1)     f32
    const float*     b2   = (const float*)d_in[4];      // (1,)        f32
    const long long* eidx = (const long long*)d_in[5];  // (2,800000)  i64
    float* out = (float*)d_out;

    // workspace: [emb bf16 : 25.6 MB][W1^T bf16 frag-order : 256 KB][scores : 3.2 MB]
    __bf16* emb16  = (__bf16*)d_ws;
    __bf16* w1t_g  = (__bf16*)((char*)d_ws + WS_W1T_OFF);
    float*  scores = (float*)((char*)d_ws + WS_SCORES_OFF);

    k_convert_w1<<<(K_TOT * DIM + 255) / 256, 256, 0, stream>>>(W1, w1t_g);
    k_convert_emb<<<((N_NODES * DIM / 4) + 255) / 256, 256, 0, stream>>>(emb, emb16);

    int n_tiles = E_EDGES / M_TILE;  // 25000
    k_edge_score<<<1024, 256, SMEM_BYTES, stream>>>(emb16, w1t_g, b1, W2, b2,
                                                    eidx, scores, n_tiles);

    k_topk<<<1, 256, 0, stream>>>(scores, eidx, out);
}
